// MyMANOLayer_42734924595230
// MI455X (gfx1250) — matrix operand order repaired and checked by reading, not compiled, not run on hardware
//
#include <hip/hip_runtime.h>
#include <hip/hip_bf16.h>
#include <hip/hip_fp16.h>

#define B_    4096
#define V_    778
#define NJ_   16
#define NC_   30
#define NS_   10
#define NF_   1538
#define NPOSE 135          // (NJ-1)*9
#define KPAD  160          // 5 x 32 (WMMA K steps)
#define NVC   2334         // V*3
#define NPAD  2336         // 146 x 16
#define JVROW 2397         // 799*3 floats per batch of jv

typedef __attribute__((ext_vector_type(16))) _Float16 v16h;
typedef __attribute__((ext_vector_type(8)))  _Float16 v8h;
typedef __attribute__((ext_vector_type(8)))  float    v8f;

// ---------------- posedirs -> padded f16 [NPAD][KPAD] ----------------
__global__ void k_conv_pd(const float* __restrict__ pd, __half* __restrict__ pdh_) {
    _Float16* pdh = reinterpret_cast<_Float16*>(pdh_);
    int tid = blockIdx.x * blockDim.x + threadIdx.x;
    if (tid >= NPAD * KPAD) return;
    int n = tid / KPAD, p = tid % KPAD;
    float v = (n < NVC && p < NPOSE) ? pd[n * NPOSE + p] : 0.f;
    pdh[tid] = (_Float16)v;
}

// ---------------- faces broadcast ----------------
__global__ void k_faces(const int* __restrict__ faces, int* __restrict__ out_faces) {
    const int per = NF_ * 3, total = B_ * per;
    for (int i = blockIdx.x * blockDim.x + threadIdx.x; i < total;
         i += gridDim.x * blockDim.x)
        out_faces[i] = faces[i % per];
}

// ---------------- pose decode + Rodrigues + pw(f16) ----------------
__global__ void k_pose(const float* __restrict__ rot,
                       const float* __restrict__ pose_params,
                       const float* __restrict__ hc,   // [30,45]
                       const float* __restrict__ hm,   // [45]
                       float* __restrict__ out_poses,  // [B,16,3]
                       float* __restrict__ wsR,        // [B,144]
                       __half* __restrict__ wsPW_) {   // [B,160] f16
    _Float16* wsPW = reinterpret_cast<_Float16*>(wsPW_);
    int tid = blockIdx.x * blockDim.x + threadIdx.x;
    if (tid >= B_ * NJ_) return;
    int b = tid >> 4, j = tid & 15;
    float p0, p1, p2;
    if (j == 0) {
        p0 = rot[b * 3 + 0]; p1 = rot[b * 3 + 1]; p2 = rot[b * 3 + 2];
        for (int t = NPOSE; t < KPAD; ++t) wsPW[(size_t)b * KPAD + t] = (_Float16)0.f;
    } else {
        int col = (j - 1) * 3;
        float a0 = hm[col], a1 = hm[col + 1], a2 = hm[col + 2];
        for (int s = 0; s < NC_; ++s) {
            float pp = pose_params[b * NC_ + s];
            a0 += pp * hc[s * 45 + col];
            a1 += pp * hc[s * 45 + col + 1];
            a2 += pp * hc[s * 45 + col + 2];
        }
        p0 = a0; p1 = a1; p2 = a2;
    }
    out_poses[(size_t)b * 48 + j * 3 + 0] = p0;
    out_poses[(size_t)b * 48 + j * 3 + 1] = p1;
    out_poses[(size_t)b * 48 + j * 3 + 2] = p2;
    // Rodrigues
    float th = sqrtf(p0 * p0 + p1 * p1 + p2 * p2 + 1e-8f);
    float inv = 1.f / th;
    float kx = p0 * inv, ky = p1 * inv, kz = p2 * inv;
    float sn = sinf(th), cs = cosf(th), C = 1.f - cs;
    float R[9];
    R[0] = cs + C * kx * kx;  R[1] = C * kx * ky - sn * kz; R[2] = C * kx * kz + sn * ky;
    R[3] = C * ky * kx + sn * kz; R[4] = cs + C * ky * ky;  R[5] = C * ky * kz - sn * kx;
    R[6] = C * kz * kx - sn * ky; R[7] = C * kz * ky + sn * kx; R[8] = cs + C * kz * kz;
    float* rdst = wsR + (size_t)b * 144 + j * 9;
#pragma unroll
    for (int e = 0; e < 9; ++e) rdst[e] = R[e];
    if (j > 0) {
        _Float16* pwdst = wsPW + (size_t)b * KPAD + (j - 1) * 9;
#pragma unroll
        for (int e = 0; e < 9; ++e) {
            float id = (e == 0 || e == 4 || e == 8) ? 1.f : 0.f;
            pwdst[e] = (_Float16)(R[e] - id);
        }
    }
}

// ---------------- shape blendshapes (K=10, VALU) -> v_shaped [B][NPAD] ----
__global__ void k_shape(const float* __restrict__ vt, const float* __restrict__ sp,
                        const float* __restrict__ sd, float* __restrict__ vs) {
    int tid = blockIdx.x * blockDim.x + threadIdx.x;
    int b = tid / NPAD, n = tid % NPAD;
    float acc = 0.f;
    if (n < NVC) {
        acc = vt[n];
        const float* sdr = sd + (size_t)n * NS_;
        const float* spb = sp + (size_t)b * NS_;
#pragma unroll
        for (int s = 0; s < NS_; ++s) acc += spb[s] * sdr[s];
    }
    vs[(size_t)b * NPAD + n] = acc;
}

// ---------------- joint regression: wave per (b,j) ----------------
__global__ void k_jreg(const float* __restrict__ Jr, const float* __restrict__ vs,
                       float* __restrict__ Jp) {
    int b = blockIdx.x;
    int j = threadIdx.x >> 5, lane = threadIdx.x & 31;
    const float* vrow = vs + (size_t)b * NPAD;
    const float* jrow = Jr + (size_t)j * V_;
    float a0 = 0.f, a1 = 0.f, a2 = 0.f;
    for (int v = lane; v < V_; v += 32) {
        float w = jrow[v];
        a0 += w * vrow[v * 3 + 0];
        a1 += w * vrow[v * 3 + 1];
        a2 += w * vrow[v * 3 + 2];
    }
#pragma unroll
    for (int off = 16; off > 0; off >>= 1) {
        a0 += __shfl_xor(a0, off, 32);
        a1 += __shfl_xor(a1, off, 32);
        a2 += __shfl_xor(a2, off, 32);
    }
    if (lane == 0) {
        Jp[(size_t)b * 48 + j * 3 + 0] = a0;
        Jp[(size_t)b * 48 + j * 3 + 1] = a1;
        Jp[(size_t)b * 48 + j * 3 + 2] = a2;
    }
}

// ---------------- FK chain + A_rel, wave per batch ----------------
__global__ void k_fk(const float* __restrict__ wsR, const float* __restrict__ JpBuf,
                     float* __restrict__ out_jv, float* __restrict__ wsArel) {
    const int lb = threadIdx.x >> 5;          // 4 batches per block
    const int lane = threadIdx.x & 31;
    const int b = blockIdx.x * 4 + lb;
    __shared__ float As[4][192];
    __shared__ float Jps[4][48];
    const int PAR[16] = {0, 0, 1, 2, 0, 4, 5, 0, 7, 8, 0, 10, 11, 0, 13, 14};
    for (int idx = lane; idx < 48; idx += 32) Jps[lb][idx] = JpBuf[(size_t)b * 48 + idx];
    __syncthreads();
    if (lane < 12) {
        int x = lane >> 2, c = lane & 3;
        As[lb][lane] = (c < 3) ? wsR[(size_t)b * 144 + x * 3 + c] : Jps[lb][x];
    }
    __syncthreads();
    for (int j = 1; j < 16; ++j) {
        int p = PAR[j];
        float val = 0.f;
        if (lane < 12) {
            int x = lane >> 2, c = lane & 3;
#pragma unroll
            for (int y = 0; y < 3; ++y) {
                float L = (c < 3) ? wsR[(size_t)b * 144 + j * 9 + y * 3 + c]
                                  : (Jps[lb][j * 3 + y] - Jps[lb][p * 3 + y]);
                val += As[lb][p * 12 + x * 4 + y] * L;
            }
            if (c == 3) val += As[lb][p * 12 + x * 4 + 3];
        }
        if (lane < 12) As[lb][j * 12 + lane] = val;
        __syncthreads();
    }
    // 16 FK joints -> output slots (fingertips filled by LBS kernel)
    for (int idx = lane; idx < 48; idx += 32) {
        int j = idx / 3, x = idx % 3;
        int slot = j + (j >= 4) + (j >= 7) + (j >= 10) + (j >= 13);
        out_jv[(size_t)b * JVROW + slot * 3 + x] = As[lb][j * 12 + x * 4 + 3];
    }
    // A_rel: subtract R_global * Jp from translation column
    for (int idx = lane; idx < 192; idx += 32) {
        int j = idx / 12, e = idx % 12, x = e >> 2, c = e & 3;
        float val = As[lb][idx];
        if (c == 3) {
            float d = 0.f;
#pragma unroll
            for (int y = 0; y < 3; ++y) d += As[lb][j * 12 + x * 4 + y] * Jps[lb][j * 3 + y];
            val -= d;
        }
        wsArel[(size_t)b * 192 + idx] = val;
    }
}

// ---------------- pose blendshape GEMM via WMMA (in-place add) ----------
// D[b,n] = sum_k pw[b,k] * PDh[n,k]  (K=160 padded, f16 x f16 -> f32)
// then v_posed = v_shaped + D, written back in place.
__global__ void k_posegemm(const __half* __restrict__ pw_, const __half* __restrict__ pdh_,
                           float* __restrict__ vsp) {
    const _Float16* pw  = reinterpret_cast<const _Float16*>(pw_);
    const _Float16* pdh = reinterpret_cast<const _Float16*>(pdh_);
    const int NT = NPAD / 16;                               // 146 N-tiles
    int wave = (blockIdx.x * blockDim.x + threadIdx.x) >> 5;
    int bT = wave / NT, nT = wave % NT;
    int lane = threadIdx.x & 31;
    int hf = lane >> 4, m = lane & 15;                      // lane half / index
    v8f acc = {0.f, 0.f, 0.f, 0.f, 0.f, 0.f, 0.f, 0.f};
    const _Float16* Arow = pw + (size_t)(bT * 16 + m) * KPAD;        // A: row m
    const _Float16* Brow = pdh + (size_t)(nT * 16 + m) * KPAD + 16 * hf; // B: col n=m
#pragma unroll
    for (int kt = 0; kt < KPAD / 32; ++kt) {
        // A fragment (contiguous K order, same as B): VGPR0-3 -> K = 16*hf + 0..7,
        //                                   VGPR4-7 -> K = 16*hf + 8..15
        const _Float16* ap = Arow + kt * 32 + 16 * hf;
        v8h alo = *reinterpret_cast<const v8h*>(ap);
        v8h ahi = *reinterpret_cast<const v8h*>(ap + 8);
        v16h afrag = __builtin_shufflevector(alo, ahi, 0, 1, 2, 3, 4, 5, 6, 7,
                                             8, 9, 10, 11, 12, 13, 14, 15);
        // B fragment (32x16): lanes 0-15 hold K=0..15, lanes 16-31 K=16..31,
        // contiguous along K for fixed column n.
        v16h bfrag = *reinterpret_cast<const v16h*>(Brow + kt * 32);
        acc = __builtin_amdgcn_wmma_f32_16x16x32_f16(false, afrag, false, bfrag,
                                                     (short)0, acc, false, false);
    }
    // D layout: VGPR r -> row (r + 8*hf), col = lane&15
    int col = nT * 16 + m;
#pragma unroll
    for (int r = 0; r < 8; ++r) {
        size_t idx = (size_t)(bT * 16 + r + 8 * hf) * NPAD + col;
        vsp[idx] = acc[r] + vsp[idx];
    }
}

// ---------------- LBS + vertex/fingertip output ----------------
__global__ void k_lbs(const float* __restrict__ wsArel, const float* __restrict__ vposed,
                      const float* __restrict__ wts, float* __restrict__ out_jv) {
    int b = blockIdx.x;
    __shared__ float As[192];
    for (int i = threadIdx.x; i < 192; i += blockDim.x)
        As[i] = wsArel[(size_t)b * 192 + i];
    __syncthreads();
    for (int v = threadIdx.x; v < V_; v += blockDim.x) {
        float T[12];
#pragma unroll
        for (int e = 0; e < 12; ++e) T[e] = 0.f;
        const float* wrow = wts + (size_t)v * NJ_;
        for (int j = 0; j < NJ_; ++j) {
            float wj = wrow[j];
#pragma unroll
            for (int e = 0; e < 12; ++e) T[e] += wj * As[j * 12 + e];
        }
        const float* vp = vposed + (size_t)b * NPAD + v * 3;
        float x = vp[0], y = vp[1], z = vp[2];
        float o0 = T[0] * x + T[1] * y + T[2] * z + T[3];
        float o1 = T[4] * x + T[5] * y + T[6] * z + T[7];
        float o2 = T[8] * x + T[9] * y + T[10] * z + T[11];
        size_t base = (size_t)b * JVROW + (size_t)(21 + v) * 3;
        out_jv[base + 0] = o0; out_jv[base + 1] = o1; out_jv[base + 2] = o2;
        int slot = (v == 320) ? 4 : (v == 443) ? 8 : (v == 672) ? 12
                 : (v == 555) ? 16 : (v == 744) ? 20 : -1;
        if (slot >= 0) {
            size_t jb = (size_t)b * JVROW + slot * 3;
            out_jv[jb + 0] = o0; out_jv[jb + 1] = o1; out_jv[jb + 2] = o2;
        }
    }
}

extern "C" void kernel_launch(void* const* d_in, const int* in_sizes, int n_in,
                              void* d_out, int out_size, void* d_ws, size_t ws_size,
                              hipStream_t stream) {
    (void)in_sizes; (void)n_in; (void)out_size; (void)ws_size;
    const float* rot        = (const float*)d_in[0];
    const float* pose_p     = (const float*)d_in[1];
    const float* shape_p    = (const float*)d_in[2];
    const float* v_template = (const float*)d_in[3];
    const float* shapedirs  = (const float*)d_in[4];
    const float* posedirs   = (const float*)d_in[5];
    const float* J_reg      = (const float*)d_in[6];
    const float* weights    = (const float*)d_in[7];
    const float* hands_comp = (const float*)d_in[8];
    const float* hands_mean = (const float*)d_in[9];
    const int*   faces      = (const int*)d_in[10];

    float* out       = (float*)d_out;
    float* out_jv    = out;                                   // [B,799,3]
    int*   out_faces = (int*)(out + (size_t)B_ * JVROW);      // [B,1538,3]
    float* out_poses = out + (size_t)B_ * (JVROW + NF_ * 3);  // [B,16,3]

    char* ws = (char*)d_ws;
    size_t off = 0;
    auto alloc = [&](size_t bytes) {
        void* p = ws + off;
        off = (off + bytes + 255) & ~(size_t)255;
        return p;
    };
    __half* wsPW   = (__half*)alloc((size_t)B_ * KPAD * sizeof(__half));
    float*  wsR    = (float*)alloc((size_t)B_ * 144 * sizeof(float));
    float*  wsVS   = (float*)alloc((size_t)B_ * NPAD * sizeof(float));  // v_shaped -> v_posed
    float*  wsJp   = (float*)alloc((size_t)B_ * 48 * sizeof(float));
    float*  wsArel = (float*)alloc((size_t)B_ * 192 * sizeof(float));
    __half* wsPDh  = (__half*)alloc((size_t)NPAD * KPAD * sizeof(__half));

    k_conv_pd<<<(NPAD * KPAD + 255) / 256, 256, 0, stream>>>(posedirs, wsPDh);
    k_faces<<<4096, 256, 0, stream>>>(faces, out_faces);
    k_pose<<<(B_ * NJ_ + 255) / 256, 256, 0, stream>>>(rot, pose_p, hands_comp,
                                                       hands_mean, out_poses, wsR, wsPW);
    k_shape<<<(B_ * NPAD) / 256, 256, 0, stream>>>(v_template, shape_p, shapedirs, wsVS);
    k_jreg<<<B_, 512, 0, stream>>>(J_reg, wsVS, wsJp);
    k_fk<<<B_ / 4, 128, 0, stream>>>(wsR, wsJp, out_jv, wsArel);
    // 256 M-tiles * 146 N-tiles = 37376 tiles, 8 waves/block -> 4672 blocks
    k_posegemm<<<(256 * (NPAD / 16)) / 8, 256, 0, stream>>>(wsPW, wsPDh, wsVS);
    k_lbs<<<B_, 256, 0, stream>>>(wsArel, wsVS, weights, out_jv);
}
